// PatchAttention_58815282151824
// MI455X (gfx1250) — matrix operand order repaired and checked by reading, not compiled, not run on hardware
//
#include <hip/hip_runtime.h>

typedef __attribute__((ext_vector_type(16))) __bf16 v16bf;
typedef __attribute__((ext_vector_type(8)))  float  v8f;
typedef int v4i __attribute__((vector_size(16)));
typedef __attribute__((address_space(1))) v4i* gv4i_p;   // global (device) ptr
typedef __attribute__((address_space(3))) v4i* lv4i_p;   // LDS ptr

#define TPI 3.14159265358979323846f

__device__ __forceinline__ unsigned short f2bf(float f) {
  union { float f; unsigned int u; } v; v.f = f;
  unsigned int u = v.u;
  return (unsigned short)((u + 0x7FFFu + ((u >> 16) & 1u)) >> 16);  // RNE
}

union FragBF { uint4 u[2]; v16bf v; };

__device__ __forceinline__ v16bf ld_frag(const unsigned short* a, const unsigned short* b) {
  FragBF f;
  f.u[0] = *reinterpret_cast<const uint4*>(a);
  f.u[1] = *reinterpret_cast<const uint4*>(b);
  return f.v;
}

__device__ __forceinline__ v8f wmma_bf16(v16bf a, v16bf b, v8f c) {
  // D = A(16x32 bf16) * B(32x16 bf16) + C(16x16 f32)
  return __builtin_amdgcn_wmma_f32_16x16x32_bf16(false, a, false, b, (short)0, c,
                                                 false, false);
}

// -------- async global->LDS copy path (CDNA5), sync fallback otherwise ------
#if defined(__HIP_DEVICE_COMPILE__) && \
    __has_builtin(__builtin_amdgcn_global_load_async_to_lds_b128)
#define USE_ASYNC_LDS 1
#else
#define USE_ASYNC_LDS 0
#endif

__device__ __forceinline__ void cp16_to_lds(unsigned short* ldst,
                                            const unsigned short* gsrc) {
#if USE_ASYNC_LDS
  __builtin_amdgcn_global_load_async_to_lds_b128((gv4i_p)gsrc, (lv4i_p)ldst, 0, 0);
#else
  *reinterpret_cast<uint4*>(ldst) = *reinterpret_cast<const uint4*>(gsrc);
#endif
}

__device__ __forceinline__ void wait_async_lds() {
#if USE_ASYNC_LDS
#if __has_builtin(__builtin_amdgcn_s_wait_asynccnt)
  __builtin_amdgcn_s_wait_asynccnt(0);
#else
  asm volatile("s_wait_asynccnt 0x0" ::: "memory");
#endif
#endif
}

// ---------------------------------------------------------------- elementwise
__global__ void cvt_bf16_kernel(const float* __restrict__ x,
                                unsigned short* __restrict__ y, int n) {
  int i = blockIdx.x * 256 + threadIdx.x;
  if (i < n) y[i] = f2bf(x[i]);
}

// W [K][N] f32 -> Wt [N][K] bf16 (coalesced via LDS tile)
__global__ __launch_bounds__(256) void transpose_bf16_kernel(
    const float* __restrict__ W, unsigned short* __restrict__ Wt, int K, int N) {
  __shared__ float tile[32][33];
  int tx = threadIdx.x & 31;
  int ty = threadIdx.x >> 5;            // 0..7
  int n0 = blockIdx.x * 32;
  int k0 = blockIdx.y * 32;
  for (int i = ty; i < 32; i += 8)
    tile[i][tx] = W[(size_t)(k0 + i) * N + n0 + tx];
  __syncthreads();
  for (int i = ty; i < 32; i += 8)
    Wt[(size_t)(n0 + i) * K + k0 + tx] = f2bf(tile[tx][i]);
}

// global eta_range = max_b(max eta_b - min eta_b), clamped to 1e-6
__global__ __launch_bounds__(1024) void eta_range_kernel(
    const float* __restrict__ pcoords, float* __restrict__ out) {
  __shared__ float smax[1024];
  __shared__ float smin[1024];
  int t = threadIdx.x;
  float range = 1e-6f;
  for (int b = 0; b < 4; ++b) {
    float e = pcoords[((size_t)b * 1024 + t) * 2];
    smax[t] = e; smin[t] = e;
    __syncthreads();
    for (int s = 512; s > 0; s >>= 1) {
      if (t < s) {
        smax[t] = fmaxf(smax[t], smax[t + s]);
        smin[t] = fminf(smin[t], smin[t + s]);
      }
      __syncthreads();
    }
    if (t == 0) range = fmaxf(range, smax[0] - smin[0]);
    __syncthreads();
  }
  if (t == 0) out[0] = range;
}

// ---------------------------------------------------------------- GEMM
// C[M=grid.x*64][N] = A[M][Kdim] * Bt[N][Kdim]^T + bias
// mode 0: bf16 out row-major; mode 1: bf16 out as Vt[b,h,d,t]; mode 2: f32 out
__global__ __launch_bounds__(256) void gemm_bf16_kernel(
    const unsigned short* __restrict__ A, const unsigned short* __restrict__ Bt,
    const float* __restrict__ bias, void* __restrict__ Out,
    int Kdim, int N, int mode) {
  __shared__ alignas(16) unsigned short sA[2][64 * 32];    // 2 x 4 KB
  __shared__ alignas(16) unsigned short sB[2][128 * 32];   // 2 x 8 KB
  const int tid = threadIdx.x;
  const int m0 = blockIdx.x * 64;
  const int n0 = blockIdx.y * 128;
  const int wave = tid >> 5;
  const int lane = tid & 31;
  const int lm = lane & 15;
  const int lh = lane >> 4;
  const int wm = (wave & 1) * 32;   // 2 waves along M
  const int wn = (wave >> 1) * 32;  // 4 waves along N

  auto stage = [&](int buf, int k0) {
    int r = tid >> 2;
    int c = (tid & 3) << 3;
    cp16_to_lds(&sA[buf][r * 32 + c],
                &A[(size_t)(m0 + r) * Kdim + k0 + c]);
#pragma unroll
    for (int it = 0; it < 2; ++it) {
      int idx = tid + it * 256;
      int r2 = idx >> 2;
      int c2 = (idx & 3) << 3;
      cp16_to_lds(&sB[buf][r2 * 32 + c2],
                  &Bt[(size_t)(n0 + r2) * Kdim + k0 + c2]);
    }
  };

  v8f acc[2][2];
#pragma unroll
  for (int i = 0; i < 2; ++i)
#pragma unroll
    for (int j = 0; j < 2; ++j)
#pragma unroll
      for (int e = 0; e < 8; ++e) acc[i][j][e] = 0.0f;

  stage(0, 0);  // prologue: fill buffer 0
  int cur = 0;
  for (int k0 = 0; k0 < Kdim; k0 += 32) {
    wait_async_lds();     // our buf[cur] async copies landed
    __syncthreads();      // everyone's landed; prior reads of buf[cur] retired
    if (k0 + 32 < Kdim) {
      stage(cur ^ 1, k0 + 32);   // overlap next tile fetch with WMMA
      if (k0 + 64 < Kdim) {      // pull tile-after-next toward L2
        __builtin_prefetch(&A[(size_t)(m0 + (tid >> 2)) * Kdim + k0 + 64], 0, 1);
        __builtin_prefetch(&Bt[(size_t)(n0 + (tid >> 2)) * Kdim + k0 + 64], 0, 1);
      }
    }

    v16bf af[2], bfr[2];
#pragma unroll
    for (int mt = 0; mt < 2; ++mt) {
      const unsigned short* ar = &sA[cur][(wm + mt * 16 + lm) * 32];
      af[mt] = ld_frag(ar + lh * 16, ar + lh * 16 + 8);
    }
#pragma unroll
    for (int nt = 0; nt < 2; ++nt) {
      const unsigned short* br = &sB[cur][(wn + nt * 16 + lm) * 32 + lh * 16];
      bfr[nt] = ld_frag(br, br + 8);
    }
#pragma unroll
    for (int mt = 0; mt < 2; ++mt)
#pragma unroll
      for (int nt = 0; nt < 2; ++nt)
        acc[mt][nt] = wmma_bf16(af[mt], bfr[nt], acc[mt][nt]);
    cur ^= 1;
  }

#pragma unroll
  for (int mt = 0; mt < 2; ++mt) {
#pragma unroll
    for (int nt = 0; nt < 2; ++nt) {
      int n = n0 + wn + nt * 16 + lm;
      float bv = bias[n];
#pragma unroll
      for (int r = 0; r < 8; ++r) {
        int m = m0 + wm + mt * 16 + r + lh * 8;
        float val = acc[mt][nt][r] + bv;
        if (mode == 2) {
          reinterpret_cast<float*>(Out)[(size_t)m * N + n] = val;
        } else if (mode == 1) {
          int bb = m >> 10, t = m & 1023;
          int hh = n >> 6, d = n & 63;
          reinterpret_cast<unsigned short*>(Out)
              [((size_t)(bb * 8 + hh) * 64 + d) * 1024 + t] = f2bf(val);
        } else {
          reinterpret_cast<unsigned short*>(Out)[(size_t)m * N + n] = f2bf(val);
        }
      }
    }
  }
}

// ---------------------------------------------------------------- attention
__device__ __forceinline__ float rpe_bias_f(float ei, float phii, float ej,
                                            float phij, float inv_range,
                                            const float* tab, int h) {
  float re = ei - ej;
  float x = phii - phij + TPI;
  x -= floorf(x * (1.0f / (2.0f * TPI))) * (2.0f * TPI);  // floor-mod -> [0,2pi)
  float rp = x - TPI;
  int eb = (int)(re * inv_range * 16.0f);
  int pb = (int)(rp * (16.0f / TPI));
  eb = eb < -16 ? -16 : (eb > 15 ? 15 : eb);
  pb = pb < -16 ? -16 : (pb > 15 ? 15 : pb);
  return tab[(eb + 16) * 8 + h] + tab[(pb + 48) * 8 + h];
}

// grid (T/64, H, B), block 128 (4 waves, 16 query rows each), flash softmax.
// All 4 waves share one (b,h): K/V tiles staged once per block via async LDS.
__global__ __launch_bounds__(128) void attn_kernel(
    const unsigned short* __restrict__ Qg, const unsigned short* __restrict__ Kg,
    const unsigned short* __restrict__ Vt, const float* __restrict__ pcoords,
    const float* __restrict__ rpe, const float* __restrict__ range_p,
    unsigned short* __restrict__ Og) {
  __shared__ float s_eta[1024];
  __shared__ float s_phi[1024];
  __shared__ float s_tab[64 * 8];
  __shared__ alignas(16) unsigned short s_p[4][16 * 32];  // per-wave P staging
  __shared__ alignas(16) unsigned short sK[32 * 64];      // 32 keys x 64 dh
  __shared__ alignas(16) unsigned short sV[64 * 32];      // 64 dh x 32 keys

  const int tid = threadIdx.x;
  const int w = tid >> 5;
  const int lane = tid & 31;
  const int lm = lane & 15;
  const int lh = lane >> 4;
  const int h = blockIdx.y;
  const int b = blockIdx.z;
  const int qrow = blockIdx.x * 64 + w * 16;

  for (int i = tid; i < 1024; i += 128) {
    s_eta[i] = pcoords[((size_t)b * 1024 + i) * 2 + 0];
    s_phi[i] = pcoords[((size_t)b * 1024 + i) * 2 + 1];
  }
  for (int i = tid; i < 512; i += 128) s_tab[i] = rpe[i];
  __syncthreads();

  const float inv_range = 1.0f / range_p[0];
  const float scale = 0.125f;  // 1/sqrt(64)

  v16bf qa[2];  // Q A-fragments, dh=64 -> two K=32 steps, resident for all keys
  {
    const unsigned short* qb =
        Qg + (size_t)(b * 1024 + qrow + lm) * 512 + h * 64;
#pragma unroll
    for (int ds = 0; ds < 2; ++ds) {
      const unsigned short* p0 = qb + ds * 32;
      qa[ds] = ld_frag(p0 + lh * 16, p0 + lh * 16 + 8);
    }
  }

  float my_eta[8], my_phi[8];
#pragma unroll
  for (int r = 0; r < 8; ++r) {
    int i = qrow + r + 8 * lh;
    my_eta[r] = s_eta[i];
    my_phi[r] = s_phi[i];
  }

  v8f o[4];
#pragma unroll
  for (int nt = 0; nt < 4; ++nt)
#pragma unroll
    for (int e = 0; e < 8; ++e) o[nt][e] = 0.0f;
  float m_run[8], l_run[8];
#pragma unroll
  for (int r = 0; r < 8; ++r) { m_run[r] = -3.0e38f; l_run[r] = 0.0f; }

  const unsigned short* Kbase = Kg + (size_t)b * 1024 * 512 + h * 64;
  const unsigned short* Vbase = Vt + (size_t)(b * 8 + h) * 64 * 1024;

  for (int kb = 0; kb < 1024; kb += 32) {
    // cooperative async staging of the shared K tile (32x128B, coalesced)
    for (int c = tid; c < 256; c += 128) {
      int row = c >> 3, col = (c & 7) << 3;
      cp16_to_lds(&sK[row * 64 + col],
                  Kbase + (size_t)(kb + row) * 512 + col);
    }
    // and the shared V tile (64 rows of 64B from pre-transposed V)
    for (int c = tid; c < 256; c += 128) {
      int d = c >> 2, kk = (c & 3) << 3;
      cp16_to_lds(&sV[d * 32 + kk], Vbase + (size_t)d * 1024 + kb + kk);
    }
    if (kb + 32 < 1024) {
      __builtin_prefetch(Kbase + (size_t)(kb + 32 + (tid >> 2)) * 512, 0, 1);
      __builtin_prefetch(Vbase + (size_t)(tid >> 1) * 1024 + kb + 32, 0, 1);
    }
    wait_async_lds();
    __syncthreads();

    v8f s0, s1;
#pragma unroll
    for (int e = 0; e < 8; ++e) { s0[e] = 0.0f; s1[e] = 0.0f; }
#pragma unroll
    for (int ds = 0; ds < 2; ++ds) {
      const unsigned short* kr0 = &sK[lm * 64 + ds * 32 + lh * 16];
      const unsigned short* kr1 = &sK[(16 + lm) * 64 + ds * 32 + lh * 16];
      s0 = wmma_bf16(qa[ds], ld_frag(kr0, kr0 + 8), s0);
      s1 = wmma_bf16(qa[ds], ld_frag(kr1, kr1 + 8), s1);
    }

    float sv0[8], sv1[8];
    float ej0 = s_eta[kb + lm], pj0 = s_phi[kb + lm];
    float ej1 = s_eta[kb + 16 + lm], pj1 = s_phi[kb + 16 + lm];
#pragma unroll
    for (int r = 0; r < 8; ++r) {
      sv0[r] = s0[r] * scale +
               rpe_bias_f(my_eta[r], my_phi[r], ej0, pj0, inv_range, s_tab, h);
      sv1[r] = s1[r] * scale +
               rpe_bias_f(my_eta[r], my_phi[r], ej1, pj1, inv_range, s_tab, h);
    }

    float alpha[8];
#pragma unroll
    for (int r = 0; r < 8; ++r) {  // row lives in one 16-lane half of the wave
      float x = fmaxf(sv0[r], sv1[r]);
#pragma unroll
      for (int off = 1; off < 16; off <<= 1) x = fmaxf(x, __shfl_xor(x, off));
      float mn = fmaxf(m_run[r], x);
      alpha[r] = __expf(m_run[r] - mn);
      m_run[r] = mn;
      float p0 = __expf(sv0[r] - mn);
      float p1 = __expf(sv1[r] - mn);
      sv0[r] = p0; sv1[r] = p1;
      float rs = p0 + p1;
#pragma unroll
      for (int off = 1; off < 16; off <<= 1) rs += __shfl_xor(rs, off);
      l_run[r] = l_run[r] * alpha[r] + rs;
    }
#pragma unroll
    for (int nt = 0; nt < 4; ++nt)
#pragma unroll
      for (int r = 0; r < 8; ++r) o[nt][r] *= alpha[r];

    // C-layout -> A-fragment layout via per-wave LDS tile
    unsigned short* pw = s_p[w];
#pragma unroll
    for (int r = 0; r < 8; ++r) {
      pw[(r + 8 * lh) * 32 + lm] = f2bf(sv0[r]);
      pw[(r + 8 * lh) * 32 + 16 + lm] = f2bf(sv1[r]);
    }
    __syncthreads();  // order LDS stores before cross-lane fragment reload
    v16bf pa = ld_frag(pw + lm * 32 + lh * 16, pw + lm * 32 + lh * 16 + 8);

#pragma unroll
    for (int nt = 0; nt < 4; ++nt) {
      const unsigned short* vr = &sV[(nt * 16 + lm) * 32 + lh * 16];
      o[nt] = wmma_bf16(pa, ld_frag(vr, vr + 8), o[nt]);
    }
    __syncthreads();  // protect sK/sV/s_p before next iteration restage
  }

  float inv_l[8];
#pragma unroll
  for (int r = 0; r < 8; ++r) inv_l[r] = 1.0f / l_run[r];
#pragma unroll
  for (int nt = 0; nt < 4; ++nt)
#pragma unroll
    for (int r = 0; r < 8; ++r) {
      int g = b * 1024 + qrow + r + 8 * lh;
      int col = h * 64 + nt * 16 + lm;
      Og[(size_t)g * 512 + col] = f2bf(o[nt][r] * inv_l[r]);
    }
}

// ---------------------------------------------------------------- launch
extern "C" void kernel_launch(void* const* d_in, const int* in_sizes, int n_in,
                              void* d_out, int out_size, void* d_ws,
                              size_t ws_size, hipStream_t stream) {
  (void)in_sizes; (void)n_in; (void)out_size; (void)ws_size;
  const float* p       = (const float*)d_in[0];
  const float* pcoords = (const float*)d_in[1];
  const float* rpe     = (const float*)d_in[2];
  const float* wq_w    = (const float*)d_in[3];
  const float* wq_b    = (const float*)d_in[4];
  const float* wk_w    = (const float*)d_in[5];
  const float* wk_b    = (const float*)d_in[6];
  const float* wv_w    = (const float*)d_in[7];
  const float* wv_b    = (const float*)d_in[8];
  const float* wo_w    = (const float*)d_in[9];
  const float* wo_b    = (const float*)d_in[10];

  const int B = 4, T = 1024, D = 512, H = 8;
  const size_t MT = (size_t)B * T;

  char* ws = (char*)d_ws;
  size_t off = 0;
  auto carve = [&](size_t bytes) {
    char* q = ws + off;
    off += (bytes + 255) & ~(size_t)255;
    return q;
  };
  unsigned short* p_bf = (unsigned short*)carve(MT * D * 2);
  unsigned short* wq_t = (unsigned short*)carve((size_t)D * D * 2);
  unsigned short* wk_t = (unsigned short*)carve((size_t)D * D * 2);
  unsigned short* wv_t = (unsigned short*)carve((size_t)D * D * 2);
  unsigned short* wo_t = (unsigned short*)carve((size_t)D * D * 2);
  unsigned short* Qb   = (unsigned short*)carve(MT * D * 2);
  unsigned short* Kb   = (unsigned short*)carve(MT * D * 2);
  unsigned short* Vtb  = (unsigned short*)carve(MT * D * 2);
  unsigned short* Ob   = (unsigned short*)carve(MT * D * 2);
  float* range_p       = (float*)carve(256);

  int n = (int)(MT * D);
  cvt_bf16_kernel<<<(n + 255) / 256, 256, 0, stream>>>(p, p_bf, n);
  dim3 tg(D / 32, D / 32);
  transpose_bf16_kernel<<<tg, 256, 0, stream>>>(wq_w, wq_t, D, D);
  transpose_bf16_kernel<<<tg, 256, 0, stream>>>(wk_w, wk_t, D, D);
  transpose_bf16_kernel<<<tg, 256, 0, stream>>>(wv_w, wv_t, D, D);
  transpose_bf16_kernel<<<tg, 256, 0, stream>>>(wo_w, wo_t, D, D);
  eta_range_kernel<<<1, 1024, 0, stream>>>(pcoords, range_p);

  dim3 gg(MT / 64, D / 128);
  gemm_bf16_kernel<<<gg, 256, 0, stream>>>(p_bf, wq_t, wq_b, Qb, D, D, 0);
  gemm_bf16_kernel<<<gg, 256, 0, stream>>>(p_bf, wk_t, wk_b, Kb, D, D, 0);
  gemm_bf16_kernel<<<gg, 256, 0, stream>>>(p_bf, wv_t, wv_b, Vtb, D, D, 1);

  dim3 ag(T / 64, H, B);
  attn_kernel<<<ag, 128, 0, stream>>>(Qb, Kb, Vtb, pcoords, rpe, range_p, Ob);

  gemm_bf16_kernel<<<gg, 256, 0, stream>>>(Ob, wo_t, wo_b, d_out, D, D, 2);
}
